// GCN_64372969832855
// MI455X (gfx1250) — matrix operand order repaired and checked by reading, not compiled, not run on hardware
//
#include <hip/hip_runtime.h>

typedef __attribute__((ext_vector_type(16))) _Float16 v16h;
typedef __attribute__((ext_vector_type(8)))  float    v8f;

#define HID 64

// ---------------- utility kernels ----------------

__global__ void zero_f32_kernel(float* __restrict__ p, size_t cnt) {
  size_t i = (size_t)blockIdx.x * blockDim.x + threadIdx.x;
  size_t stride = (size_t)gridDim.x * blockDim.x;
  for (; i < cnt; i += stride) p[i] = 0.0f;
}

__global__ void zero_i32_kernel(int* __restrict__ p, int cnt) {
  int i = blockIdx.x * blockDim.x + threadIdx.x;
  if (i < cnt) p[i] = 0;
}

__global__ void count_deg_kernel(const int* __restrict__ src, int E, int* __restrict__ deg) {
  int i = blockIdx.x * blockDim.x + threadIdx.x;
  if (i < E) atomicAdd(&deg[src[i]], 1);
}

__global__ void make_dis_kernel(const int* __restrict__ deg, int n, float* __restrict__ dis) {
  int i = blockIdx.x * blockDim.x + threadIdx.x;
  if (i < n) dis[i] = rsqrtf((float)(deg[i] + 1));  // +1 self loop; deg>=1 always
}

// ---------------- BatchNorm statistics ----------------
// cols divides 256 (128 or 64), so each thread only ever touches column
// (threadIdx.x % cols) under the grid-stride loop.
__global__ void col_stats_kernel(const float* __restrict__ X, int rows, int cols,
                                 float* __restrict__ sums /* [2*cols], pre-zeroed */) {
  __shared__ float ssum[256];
  __shared__ float ssq[256];
  const int t = threadIdx.x;
  float s = 0.f, q = 0.f;
  const size_t total = (size_t)rows * cols;
  for (size_t i = (size_t)blockIdx.x * blockDim.x + t; i < total;
       i += (size_t)gridDim.x * blockDim.x) {
    float v = X[i];
    s += v; q += v * v;
  }
  ssum[t] = s; ssq[t] = q;
  __syncthreads();
  for (int half = 128; half >= cols; half >>= 1) {
    if (t < half) { ssum[t] += ssum[t + half]; ssq[t] += ssq[t + half]; }
    __syncthreads();
  }
  if (t < cols) {
    unsafeAtomicAdd(&sums[t], ssum[t]);
    unsafeAtomicAdd(&sums[cols + t], ssq[t]);
  }
}

__global__ void bn_finalize_kernel(const float* __restrict__ sums,
                                   const float* __restrict__ gamma,
                                   const float* __restrict__ beta,
                                   int cols, float inv_rows,
                                   float* __restrict__ scale, float* __restrict__ shift) {
  int t = threadIdx.x;
  if (t < cols) {
    float mean = sums[t] * inv_rows;
    float var  = sums[cols + t] * inv_rows - mean * mean;   // biased variance
    float sc = gamma[t] * rsqrtf(var + 1e-5f);
    scale[t] = sc;
    shift[t] = beta[t] - mean * sc;
  }
}

// Fold BN into GEMM operands. Produces TRANSPOSED f16 weights:
//   Wt[n*K + k] = scale[k] * W[k][n]
// so a WMMA B-fragment (16 K-values at fixed column n) is contiguous.
// biasOut[j] = sum_k shift[k]*W[k][j] (+ baseBias). Single-block launch.
__global__ void fold_w_kernel(const float* __restrict__ W,
                              const float* __restrict__ scale,
                              const float* __restrict__ shift,
                              const float* __restrict__ baseBias,  // may be null
                              int K, _Float16* __restrict__ Wt,
                              float* __restrict__ biasOut) {
  const int t = threadIdx.x;
  const int total = K * HID;
  for (int i = t; i < total; i += blockDim.x) {
    int nn = i / K;
    int kk = i - nn * K;
    Wt[i] = (_Float16)(scale[kk] * W[kk * HID + nn]);
  }
  if (t < HID) {
    float acc = baseBias ? baseBias[t] : 0.0f;
    for (int k = 0; k < K; ++k) acc += shift[k] * W[k * HID + t];
    biasOut[t] = acc;
  }
}

// ---------------- WMMA GEMM: Out[rows x 64] = A[rows x K] * W + bias ---------
// Wt is transposed [64 x K] f16. One wave computes a 16x64 tile with
// 4 x v_wmma_f32_16x16x32_f16 per K-step; K templated so the loop fully
// unrolls. Wt staged in LDS (padded rows -> bank-conflict-free ds_load_b128).
template <int K>
__global__ void gemm_wmma_k_kernel(const float* __restrict__ A,
                                   const _Float16* __restrict__ Wt,
                                   const float* __restrict__ bias,
                                   float* __restrict__ Out,
                                   int rows, int do_relu) {
  constexpr int KP = K + 8;                       // pad 8 halfs (16B) per row
  __shared__ _Float16 sW[HID * KP];

  // Cooperative copy Wt -> LDS (uint4 = 8 halfs per transfer)
  const int t = threadIdx.x;
  constexpr int V4_PER_ROW = K / 8;
  constexpr int NV4 = HID * V4_PER_ROW;
  const uint4* gsrc = (const uint4*)Wt;
  for (int i = t; i < NV4; i += 256) {
    int row = i / V4_PER_ROW;
    int off = i - row * V4_PER_ROW;
    *(uint4*)(sW + row * KP + off * 8) = gsrc[i];
  }
  __syncthreads();

  const int lane  = t & 31;
  const int wave  = t >> 5;
  const int mtile = blockIdx.x * 8 + wave;
  if (mtile * 16 >= rows) return;                 // wave-uniform; EXEC stays full
  const int hi   = lane >> 4;                     // lane half: 0 or 1
  const int lrow = lane & 15;
  int m = mtile * 16 + lrow;
  if (m >= rows) m = rows - 1;                    // clamp loads; stores guarded

  const float4* arow4 = (const float4*)(A + (size_t)m * K);
  v8f c0 = {}, c1 = {}, c2 = {}, c3 = {};

#pragma unroll
  for (int k0 = 0; k0 < K; k0 += 32) {
    // A fragment: halves 0..15 -> K=k0+hi*16+j (contiguous, matches B).
    // Four b128 loads.
    float4 f0 = arow4[(k0 >> 2) + hi * 4 + 0];
    float4 f1 = arow4[(k0 >> 2) + hi * 4 + 1];
    float4 f2 = arow4[(k0 >> 2) + hi * 4 + 2];
    float4 f3 = arow4[(k0 >> 2) + hi * 4 + 3];
    v16h a;
    a[0] = (_Float16)f0.x; a[1] = (_Float16)f0.y; a[2]  = (_Float16)f0.z; a[3]  = (_Float16)f0.w;
    a[4] = (_Float16)f1.x; a[5] = (_Float16)f1.y; a[6]  = (_Float16)f1.z; a[7]  = (_Float16)f1.w;
    a[8] = (_Float16)f2.x; a[9] = (_Float16)f2.y; a[10] = (_Float16)f2.z; a[11] = (_Float16)f2.w;
    a[12] = (_Float16)f3.x; a[13] = (_Float16)f3.y; a[14] = (_Float16)f3.z; a[15] = (_Float16)f3.w;

    // B fragments: lane holds K=kb..kb+15 at column n (contiguous in sW row)
    const int kb = k0 + hi * 16;
    v16h b0, b1, b2, b3;
    {
      const _Float16* wp0 = sW + (lrow +  0) * KP + kb;
      const _Float16* wp1 = sW + (lrow + 16) * KP + kb;
      const _Float16* wp2 = sW + (lrow + 32) * KP + kb;
      const _Float16* wp3 = sW + (lrow + 48) * KP + kb;
      ((uint4*)&b0)[0] = *(const uint4*)wp0; ((uint4*)&b0)[1] = *(const uint4*)(wp0 + 8);
      ((uint4*)&b1)[0] = *(const uint4*)wp1; ((uint4*)&b1)[1] = *(const uint4*)(wp1 + 8);
      ((uint4*)&b2)[0] = *(const uint4*)wp2; ((uint4*)&b2)[1] = *(const uint4*)(wp2 + 8);
      ((uint4*)&b3)[0] = *(const uint4*)wp3; ((uint4*)&b3)[1] = *(const uint4*)(wp3 + 8);
    }
    c0 = __builtin_amdgcn_wmma_f32_16x16x32_f16(false, a, false, b0, (short)0, c0, false, false);
    c1 = __builtin_amdgcn_wmma_f32_16x16x32_f16(false, a, false, b1, (short)0, c1, false, false);
    c2 = __builtin_amdgcn_wmma_f32_16x16x32_f16(false, a, false, b2, (short)0, c2, false, false);
    c3 = __builtin_amdgcn_wmma_f32_16x16x32_f16(false, a, false, b3, (short)0, c3, false, false);
  }

  // Epilogue: accumulator element v lives at row v+hi*8, col lrow (+16*nt)
  const float bb0 = bias[lrow], bb1 = bias[lrow + 16];
  const float bb2 = bias[lrow + 32], bb3 = bias[lrow + 48];
#pragma unroll
  for (int v = 0; v < 8; ++v) {
    int r = mtile * 16 + v + hi * 8;
    if (r < rows) {
      float* orow = Out + (size_t)r * HID + lrow;
      float e0 = c0[v] + bb0, e1 = c1[v] + bb1, e2 = c2[v] + bb2, e3 = c3[v] + bb3;
      if (do_relu) {
        e0 = fmaxf(e0, 0.f); e1 = fmaxf(e1, 0.f);
        e2 = fmaxf(e2, 0.f); e3 = fmaxf(e3, 0.f);
      }
      orow[0] = e0; orow[16] = e1; orow[32] = e2; orow[48] = e3;
    }
  }
}

// Generic runtime-K fallback (K multiple of 32): same layout, global b128 reads.
__global__ void gemm_wmma_gen_kernel(const float* __restrict__ A,
                                     const _Float16* __restrict__ Wt,
                                     const float* __restrict__ bias,
                                     float* __restrict__ Out,
                                     int rows, int K, int do_relu) {
  const int lane  = threadIdx.x & 31;
  const int wave  = threadIdx.x >> 5;
  const int mtile = blockIdx.x * 8 + wave;
  if (mtile * 16 >= rows) return;
  const int hi   = lane >> 4;
  const int lrow = lane & 15;
  int m = mtile * 16 + lrow;
  if (m >= rows) m = rows - 1;
  const float4* arow4 = (const float4*)(A + (size_t)m * K);
  v8f c0 = {}, c1 = {}, c2 = {}, c3 = {};
  for (int k0 = 0; k0 < K; k0 += 32) {
    float4 f0 = arow4[(k0 >> 2) + hi * 4 + 0];
    float4 f1 = arow4[(k0 >> 2) + hi * 4 + 1];
    float4 f2 = arow4[(k0 >> 2) + hi * 4 + 2];
    float4 f3 = arow4[(k0 >> 2) + hi * 4 + 3];
    v16h a;
    a[0] = (_Float16)f0.x; a[1] = (_Float16)f0.y; a[2]  = (_Float16)f0.z; a[3]  = (_Float16)f0.w;
    a[4] = (_Float16)f1.x; a[5] = (_Float16)f1.y; a[6]  = (_Float16)f1.z; a[7]  = (_Float16)f1.w;
    a[8] = (_Float16)f2.x; a[9] = (_Float16)f2.y; a[10] = (_Float16)f2.z; a[11] = (_Float16)f2.w;
    a[12] = (_Float16)f3.x; a[13] = (_Float16)f3.y; a[14] = (_Float16)f3.z; a[15] = (_Float16)f3.w;
    const int kb = k0 + hi * 16;
    v16h b0, b1, b2, b3;
    const _Float16* wp0 = Wt + (size_t)(lrow +  0) * K + kb;
    const _Float16* wp1 = Wt + (size_t)(lrow + 16) * K + kb;
    const _Float16* wp2 = Wt + (size_t)(lrow + 32) * K + kb;
    const _Float16* wp3 = Wt + (size_t)(lrow + 48) * K + kb;
    ((uint4*)&b0)[0] = *(const uint4*)wp0; ((uint4*)&b0)[1] = *(const uint4*)(wp0 + 8);
    ((uint4*)&b1)[0] = *(const uint4*)wp1; ((uint4*)&b1)[1] = *(const uint4*)(wp1 + 8);
    ((uint4*)&b2)[0] = *(const uint4*)wp2; ((uint4*)&b2)[1] = *(const uint4*)(wp2 + 8);
    ((uint4*)&b3)[0] = *(const uint4*)wp3; ((uint4*)&b3)[1] = *(const uint4*)(wp3 + 8);
    c0 = __builtin_amdgcn_wmma_f32_16x16x32_f16(false, a, false, b0, (short)0, c0, false, false);
    c1 = __builtin_amdgcn_wmma_f32_16x16x32_f16(false, a, false, b1, (short)0, c1, false, false);
    c2 = __builtin_amdgcn_wmma_f32_16x16x32_f16(false, a, false, b2, (short)0, c2, false, false);
    c3 = __builtin_amdgcn_wmma_f32_16x16x32_f16(false, a, false, b3, (short)0, c3, false, false);
  }
  const float bb0 = bias[lrow], bb1 = bias[lrow + 16];
  const float bb2 = bias[lrow + 32], bb3 = bias[lrow + 48];
#pragma unroll
  for (int v = 0; v < 8; ++v) {
    int r = mtile * 16 + v + hi * 8;
    if (r < rows) {
      float* orow = Out + (size_t)r * HID + lrow;
      float e0 = c0[v] + bb0, e1 = c1[v] + bb1, e2 = c2[v] + bb2, e3 = c3[v] + bb3;
      if (do_relu) {
        e0 = fmaxf(e0, 0.f); e1 = fmaxf(e1, 0.f);
        e2 = fmaxf(e2, 0.f); e3 = fmaxf(e3, 0.f);
      }
      orow[0] = e0; orow[16] = e1; orow[32] = e2; orow[48] = e3;
    }
  }
}

// ---------------- edge scatter: acc[dst] += dis[src]*dis[dst]*hw[src] --------
// 2 features per thread: halves index/dis loads per pair of f32 atomics.
__global__ void edge_scatter_kernel(const float* __restrict__ hw,
                                    const float* __restrict__ dis,
                                    const int* __restrict__ src,
                                    const int* __restrict__ dst,
                                    float* __restrict__ acc, long long totalE32) {
  long long tid = (long long)blockIdx.x * blockDim.x + threadIdx.x;
  if (tid >= totalE32) return;
  int e = (int)(tid >> 5);
  int f = (int)((tid & 31) << 1);
  int s = src[e], d = dst[e];
  float norm = dis[s] * dis[d];
  float2 hv = *(const float2*)&hw[(size_t)s * HID + f];
  float* ap = &acc[(size_t)d * HID + f];
  unsafeAtomicAdd(ap + 0, norm * hv.x);
  unsafeAtomicAdd(ap + 1, norm * hv.y);
}

// self-loop + bias + relu (float4 streaming): h = relu(acc + dis^2*hw + b)
__global__ void self_bias_relu_kernel(const float4* __restrict__ acc,
                                      const float4* __restrict__ hw,
                                      const float* __restrict__ dis,
                                      const float4* __restrict__ b4,
                                      float4* __restrict__ hout, long long totalNF4) {
  long long tid = (long long)blockIdx.x * blockDim.x + threadIdx.x;
  if (tid >= totalNF4) return;
  int i  = (int)(tid >> 4);            // 16 float4 per 64-wide row
  int f4 = (int)(tid & 15);
  float d = dis[i];
  float d2 = d * d;
  float4 a = acc[tid], h = hw[tid], bb = b4[f4];
  float4 r;
  r.x = fmaxf(a.x + d2 * h.x + bb.x, 0.f);
  r.y = fmaxf(a.y + d2 * h.y + bb.y, 0.f);
  r.z = fmaxf(a.z + d2 * h.z + bb.z, 0.f);
  r.w = fmaxf(a.w + d2 * h.w + bb.w, 0.f);
  hout[tid] = r;
}

// global_add_pool: 2 features per thread
__global__ void pool_kernel(const float* __restrict__ h, const int* __restrict__ batch,
                            float* __restrict__ out, long long totalN32) {
  long long tid = (long long)blockIdx.x * blockDim.x + threadIdx.x;
  if (tid >= totalN32) return;
  int i = (int)(tid >> 5);
  int f = (int)((tid & 31) << 1);
  int g = batch[i];
  float2 hv = *(const float2*)&h[(size_t)i * HID + f];
  float* op = &out[(size_t)g * HID + f];
  unsafeAtomicAdd(op + 0, hv.x);
  unsafeAtomicAdd(op + 1, hv.y);
}

// ---------------- orchestration ----------------

extern "C" void kernel_launch(void* const* d_in, const int* in_sizes, int n_in,
                              void* d_out, int out_size, void* d_ws, size_t ws_size,
                              hipStream_t stream) {
  const float* x     = (const float*)d_in[0];
  const int*   eidx  = (const int*)  d_in[1];
  const int*   batch = (const int*)  d_in[2];
  const float* bnfg  = (const float*)d_in[3];
  const float* bnfb  = (const float*)d_in[4];
  const float* Wfeat = (const float*)d_in[5];
  const float* bfeat = (const float*)d_in[6];
  const float* bng   = (const float*)d_in[7];
  const float* bnb   = (const float*)d_in[8];
  const float* Ws    = (const float*)d_in[9];
  const float* bs    = (const float*)d_in[10];

  const int n = in_sizes[2];                // N nodes
  const int E = in_sizes[1] / 2;            // edges
  const int F = in_sizes[0] / n;            // 128 input features
  const int Lnum = in_sizes[10] / HID;      // layers (bs is [L,64])
  const int* srcp = eidx;
  const int* dstp = eidx + E;

  char* wsp = (char*)d_ws;
  auto carve = [&](size_t bytes) -> void* {
    void* p = (void*)wsp;
    wsp += (bytes + 255) & ~(size_t)255;
    return p;
  };
  float*    bufH   = (float*)carve((size_t)n * HID * 4);
  float*    bufHW  = (float*)carve((size_t)n * HID * 4);
  float*    bufAcc = (float*)carve((size_t)n * HID * 4);
  float*    dis    = (float*)carve((size_t)n * 4);
  int*      deg    = (int*)  carve((size_t)n * 4);
  float*    sums   = (float*)carve(256 * 4);
  float*    scale  = (float*)carve(128 * 4);
  float*    shift  = (float*)carve(128 * 4);
  _Float16* Wt     = (_Float16*)carve((size_t)F * HID * 2);
  float*    biasf  = (float*)carve(64 * 4);

  const int TB = 256;
  const long long totalNF  = (long long)n * HID;
  const long long totalNF4 = totalNF >> 2;
  const long long totalE32 = (long long)E * 32;
  const long long totalN32 = (long long)n * 32;
  const unsigned nBlocks   = (unsigned)((n + TB - 1) / TB);
  const unsigned nfBlocks  = (unsigned)((totalNF + TB - 1) / TB);
  const unsigned nf4Blocks = (unsigned)((totalNF4 + TB - 1) / TB);
  const unsigned e32Blocks = (unsigned)((totalE32 + TB - 1) / TB);
  const unsigned n32Blocks = (unsigned)((totalN32 + TB - 1) / TB);
  const unsigned mBlocks   = (unsigned)((n + 127) / 128);  // 8 wave-tiles/block

  // Degrees and symmetric-norm factors (layer invariant)
  zero_i32_kernel<<<nBlocks, TB, 0, stream>>>(deg, n);
  count_deg_kernel<<<(unsigned)((E + TB - 1) / TB), TB, 0, stream>>>(srcp, E, deg);
  make_dis_kernel<<<nBlocks, TB, 0, stream>>>(deg, n, dis);

  // Feature stage: h = relu(BN(x) @ W_feat + b_feat), BN folded into W/bias
  zero_f32_kernel<<<1, TB, 0, stream>>>(sums, 256);
  col_stats_kernel<<<512, TB, 0, stream>>>(x, n, F, sums);
  bn_finalize_kernel<<<1, 128, 0, stream>>>(sums, bnfg, bnfb, F, 1.0f / (float)n, scale, shift);
  fold_w_kernel<<<1, TB, 0, stream>>>(Wfeat, scale, shift, bfeat, F, Wt, biasf);
  if (F == 128) {
    gemm_wmma_k_kernel<128><<<mBlocks, TB, 0, stream>>>(x, Wt, biasf, bufH, n, 1);
  } else if (F == 64) {
    gemm_wmma_k_kernel<64><<<mBlocks, TB, 0, stream>>>(x, Wt, biasf, bufH, n, 1);
  } else {
    gemm_wmma_gen_kernel<<<mBlocks, TB, 0, stream>>>(x, Wt, biasf, bufH, n, F, 1);
  }

  // GCN layers
  for (int l = 0; l < Lnum; ++l) {
    zero_f32_kernel<<<1, TB, 0, stream>>>(sums, 256);
    col_stats_kernel<<<512, TB, 0, stream>>>(bufH, n, HID, sums);
    bn_finalize_kernel<<<1, 128, 0, stream>>>(sums, bng + l * HID, bnb + l * HID,
                                              HID, 1.0f / (float)n, scale, shift);
    fold_w_kernel<<<1, TB, 0, stream>>>(Ws + (size_t)l * HID * HID, scale, shift,
                                        nullptr, HID, Wt, biasf);
    // hw = BN(h) @ W_l  (bias = shift . W_l, exact; no relu here)
    gemm_wmma_k_kernel<64><<<mBlocks, TB, 0, stream>>>(bufH, Wt, biasf, bufHW, n, 0);
    // acc[dst] += dis[src]*dis[dst]*hw[src]
    zero_f32_kernel<<<nfBlocks, TB, 0, stream>>>(bufAcc, (size_t)totalNF);
    edge_scatter_kernel<<<e32Blocks, TB, 0, stream>>>(bufHW, dis, srcp, dstp, bufAcc, totalE32);
    // h = relu(acc + dis^2*hw + b_l)
    self_bias_relu_kernel<<<nf4Blocks, TB, 0, stream>>>(
        (const float4*)bufAcc, (const float4*)bufHW, dis,
        (const float4*)(bs + l * HID), (float4*)bufH, totalNF4);
  }

  // graph_emb = segment_sum(h, batch)
  zero_f32_kernel<<<(unsigned)((out_size + TB - 1) / TB), TB, 0, stream>>>(
      (float*)d_out, (size_t)out_size);
  pool_kernel<<<n32Blocks, TB, 0, stream>>>(bufH, batch, (float*)d_out, totalN32);
}
